// SaConvLSTM2DCell_27221502722160
// MI455X (gfx1250) — matrix operand order repaired and checked by reading, not compiled, not run on hardware
//
#include <hip/hip_runtime.h>
#include <hip/hip_bf16.h>
#include <math.h>

// ---------------- problem constants ----------------
#define Bm 8
#define Hm 48
#define Wm 48
#define Cm 96
#define Fm 96
#define HWm (Hm * Wm)            // 2304
#define Mm (Bm * HWm)            // 18432 pixels
#define KKm (9 * (Cm + Fm))      // 1728 reduction
#define NGm (4 * Fm)             // 384 gate channels
#define Sm (Bm * HWm * Fm)       // 1769472 elems per output tensor
#define NKC (KKm / 32)           // 54 K-chunks
#define NNT (NGm / 16)           // 24 column tiles

typedef __attribute__((ext_vector_type(16))) __bf16 v16bf;
typedef __attribute__((ext_vector_type(8)))  __bf16 v8bf;
typedef __attribute__((ext_vector_type(8)))  float  v8f;

__device__ __forceinline__ float hsig(float x) {
    return fminf(fmaxf(0.2f * x + 0.5f, 0.0f), 1.0f);
}

// ---------------- K0a: pack x || h -> bf16 (B,H,W,192) ----------------
__global__ void pack_xh(const float* __restrict__ x, const float* __restrict__ h,
                        __bf16* __restrict__ xh) {
    int idx = blockIdx.x * blockDim.x + threadIdx.x;
    if (idx >= Mm * 192) return;
    int p = idx / 192, ch = idx % 192;
    float v = (ch < Cm) ? x[p * Cm + ch] : h[p * Fm + (ch - Cm)];
    xh[idx] = (__bf16)v;
}

// ---------------- K0b: pack weights -> bf16 in WMMA B-fragment order ----------------
// Layout: [kchunk 54][ntile 24][lane 32][j 16]
//   lane = half*16 + col ; element j -> k = kc*32 + 16*half + j, n = nt*16 + col
__global__ void pack_w(const float* __restrict__ wk, const float* __restrict__ wr,
                       __bf16* __restrict__ wqf) {
    int idx = blockIdx.x * blockDim.x + threadIdx.x;
    if (idx >= KKm * NGm) return;
    int kc   = idx / (NNT * 512);
    int r    = idx % (NNT * 512);
    int nt   = r / 512;
    int q    = r % 512;
    int lane = q / 16;
    int j    = q % 16;
    int half = lane >> 4, col = lane & 15;
    int k = kc * 32 + 16 * half + j;
    int n = nt * 16 + col;
    int tap = k / 192, ch = k % 192;
    float v = (ch < Cm) ? wk[(tap * Cm + ch) * NGm + n]
                        : wr[(tap * Fm + (ch - Cm)) * NGm + n];
    wqf[idx] = (__bf16)v;
}

// ---------------- K1: implicit-GEMM gates via WMMA bf16, software-pipelined ----------------
// grid (288, 2), block 256 (8 waves). Tile M=64, N=192.
// Wave w: mSub = w%4 (16 rows), nHalf = w/4 (96 cols) -> 6 16x16 accum tiles.
__global__ void __launch_bounds__(256)
gates_wmma(const __bf16* __restrict__ xh,
           const __bf16* __restrict__ wqf,
           float* __restrict__ g) {
    __shared__ alignas(16) __bf16 Alds[64][40];      // 64 pixels x 32 K (pad 8)
    __shared__ alignas(16) __bf16 Blds[12 * 512];    // 12 ntiles in fragment order

    const int tid   = threadIdx.x;
    const int lane  = tid & 31;
    const int wave  = tid >> 5;
    const int mSub  = wave & 3;
    const int nHalf = wave >> 2;
    const int half  = lane >> 4;
    const int l16   = lane & 15;

    const int mbase = blockIdx.x * 64;
    const int nt0   = blockIdx.y * 12;               // first column tile
    const int nbase = nt0 * 16;

    // A staging role: pixel p = tid/4, 8-channel quarter q = tid%4
    const int ap = tid >> 2;
    const int aq = (tid & 3) * 8;
    const int mg  = mbase + ap;
    const int bb  = mg / HWm;
    const int rem = mg % HWm;
    const int yy0 = rem / Wm;
    const int xx0 = rem % Wm;

    v8f acc[6];
#pragma unroll
    for (int t = 0; t < 6; ++t) acc[t] = (v8f){0,0,0,0,0,0,0,0};

    float4 aReg, bReg0, bReg1, bReg2;
    bool   aInb = false;
    const float4 zero4 = make_float4(0.f, 0.f, 0.f, 0.f);

    // ---- fetch chunk s into registers (global) ----
    auto fetch = [&](int s) {
        const int tap = s / 6, c = s % 6;
        const int dy = tap / 3 - 1, dx = tap % 3 - 1;
        const int yy = yy0 + dy, xx = xx0 + dx;
        aInb = ((unsigned)yy < (unsigned)Hm) & ((unsigned)xx < (unsigned)Wm);
        const int yyc = min(max(yy, 0), Hm - 1);
        const int xxc = min(max(xx, 0), Wm - 1);
        const float4* apg = (const float4*)(xh +
            (((size_t)(bb * Hm + yyc) * Wm + xxc) * 192 + c * 32 + aq));
        aReg = *apg;
        const float4* bpg = (const float4*)(wqf + ((size_t)s * NNT + nt0) * 512) + tid * 3;
        bReg0 = bpg[0]; bReg1 = bpg[1]; bReg2 = bpg[2];
    };
    // ---- commit registers into LDS ----
    auto commit = [&]() {
        *(float4*)(&Alds[ap][aq]) = aInb ? aReg : zero4;
        float4* bl = (float4*)Blds + tid * 3;
        bl[0] = bReg0; bl[1] = bReg1; bl[2] = bReg2;
    };

    fetch(0);
    commit();

    for (int s = 0; s < NKC; ++s) {
        __syncthreads();
        if (s + 1 < NKC) fetch(s + 1);   // global loads overlap the WMMAs below

        // ---- A fragment (16-bit 16x32 layout): row = mSub*16 + l16 ----
        v16bf afrag;
        {
            const __bf16* arow = &Alds[mSub * 16 + l16][16 * half];
#pragma unroll
            for (int j = 0; j < 8; ++j) afrag[j]     = arow[j];
#pragma unroll
            for (int j = 0; j < 8; ++j) afrag[8 + j] = arow[8 + j];
        }
        // ---- 6 B fragments: contiguous 32B per lane in fragment order ----
        v16bf bfrag[6];
#pragma unroll
        for (int t = 0; t < 6; ++t) {
            const __bf16* bp = Blds + ((nHalf * 6 + t) * 32 + lane) * 16;
#pragma unroll
            for (int j = 0; j < 16; ++j) bfrag[t][j] = bp[j];
        }
#pragma unroll
        for (int t = 0; t < 6; ++t) {
            acc[t] = __builtin_amdgcn_wmma_f32_16x16x32_bf16(
                false, afrag, false, bfrag[t], (short)0, acc[t], false, false);
        }

        __syncthreads();
        if (s + 1 < NKC) commit();
    }

    // ---- store C (f32 16x16 C layout: VGPR v -> M = v + 8*half, N = l16) ----
#pragma unroll
    for (int t = 0; t < 6; ++t) {
        const int col = nbase + nHalf * 96 + t * 16 + l16;
#pragma unroll
        for (int v = 0; v < 8; ++v) {
            const int row = mbase + mSub * 16 + v + 8 * half;
            g[(size_t)row * NGm + col] = acc[t][v];
        }
    }
}

// ---------------- K2: LSTM pointwise: i,f,c,o,h ----------------
__global__ void lstm_pointwise(const float* __restrict__ g,
                               const float* __restrict__ bias,
                               const float* __restrict__ c_tm1,
                               float* __restrict__ out,   // c -> out[Sm + idx]
                               float* __restrict__ hbuf) {
    int idx = blockIdx.x * blockDim.x + threadIdx.x;
    if (idx >= Sm) return;
    int p = idx / Fm, f = idx % Fm;
    size_t gb = (size_t)p * NGm;
    float gi = g[gb + f]          + bias[f];
    float gf = g[gb + Fm + f]     + bias[Fm + f];
    float gc = g[gb + 2*Fm + f]   + bias[2*Fm + f];
    float go = g[gb + 3*Fm + f]   + bias[3*Fm + f];
    float i  = hsig(gi);
    float ff = hsig(gf);
    float c  = ff * c_tm1[idx] + i * tanhf(gc);
    float o  = hsig(go);
    out[Sm + idx] = c;
    hbuf[idx]     = o * tanhf(c);
}

// ---------------- K3: five 1-channel 1x1 projections ----------------
__global__ void qkv_proj(const float* __restrict__ hbuf,
                         const float* __restrict__ m_tm1,
                         const float* __restrict__ sa_kernel,
                         float* __restrict__ qkv) {
    int idx = blockIdx.x * blockDim.x + threadIdx.x;   // proj * M + pixel
    if (idx >= 5 * Mm) return;
    int proj = idx / Mm, p = idx % Mm;
    const float* src = (proj < 3) ? hbuf : m_tm1;
    float s = 0.f;
    const float* row = src + (size_t)p * Fm;
#pragma unroll 4
    for (int f = 0; f < Fm; ++f) s += row[f] * sa_kernel[f * 5 + proj];
    qkv[(size_t)proj * Mm + p] = s;
}

// ---------------- K4: rank-1-logit softmax attention (streaming, no NxN) ----------------
// grid (B, N/64), block 64
__global__ void sam_attention(const float* __restrict__ qkv, float* __restrict__ z2) {
    __shared__ float kh[HWm], vh[HWm], km[HWm], vm[HWm];
    const int b   = blockIdx.x;
    const int i   = blockIdx.y * 64 + threadIdx.x;
    const float* qh  = qkv + 0 * Mm + (size_t)b * HWm;
    const float* khg = qkv + 1 * Mm + (size_t)b * HWm;
    const float* vhg = qkv + 2 * Mm + (size_t)b * HWm;
    const float* kmg = qkv + 3 * Mm + (size_t)b * HWm;
    const float* vmg = qkv + 4 * Mm + (size_t)b * HWm;
    for (int j = threadIdx.x; j < HWm; j += 64) {
        kh[j] = khg[j]; vh[j] = vhg[j]; km[j] = kmg[j]; vm[j] = vmg[j];
    }
    __syncthreads();

    const float qi = qh[i];
    // ---- h branch ----
    float mx = -INFINITY;
    for (int j = 0; j < HWm; ++j) mx = fmaxf(mx, qi * kh[j]);
    float den = 0.f, num = 0.f;
    for (int j = 0; j < HWm; ++j) {
        float e = __expf(qi * kh[j] - mx);
        den += e; num += e * vh[j];
    }
    float zh = num / den;
    // ---- m branch ----
    mx = -INFINITY;
    for (int j = 0; j < HWm; ++j) mx = fmaxf(mx, qi * km[j]);
    den = 0.f; num = 0.f;
    for (int j = 0; j < HWm; ++j) {
        float e = __expf(qi * km[j] - mx);
        den += e; num += e * vm[j];
    }
    float zm = num / den;

    z2[((size_t)b * HWm + i) * 2 + 0] = zh;
    z2[((size_t)b * HWm + i) * 2 + 1] = zm;
}

// ---------------- K5: conv2d(z2, kernel_z) : 3x3, 2 -> 96 ----------------
__global__ void conv_z(const float* __restrict__ z2,
                       const float* __restrict__ kz,
                       float* __restrict__ z) {
    int idx = blockIdx.x * blockDim.x + threadIdx.x;
    if (idx >= Sm) return;
    int f = idx % Fm, p = idx / Fm;
    int b = p / HWm, rem = p % HWm, y = rem / Wm, x = rem % Wm;
    float s = 0.f;
#pragma unroll
    for (int ky = 0; ky < 3; ++ky) {
        int yy = y + ky - 1;
        if (yy < 0 || yy >= Hm) continue;
#pragma unroll
        for (int kx = 0; kx < 3; ++kx) {
            int xx = x + kx - 1;
            if (xx < 0 || xx >= Wm) continue;
            size_t pp = ((size_t)(b * Hm + yy) * Wm + xx) * 2;
            int wb = (ky * 3 + kx) * 2;
            s += z2[pp + 0] * kz[(wb + 0) * Fm + f];
            s += z2[pp + 1] * kz[(wb + 1) * Fm + f];
        }
    }
    z[idx] = s;
}

// ---------------- K6: fused depthwise convs + memory update ----------------
__global__ void sam_update(const float* __restrict__ z,
                           const float* __restrict__ hbuf,
                           const float* __restrict__ dw,
                           const float* __restrict__ sa_bias,
                           const float* __restrict__ m_tm1,
                           float* __restrict__ out) {
    int idx = blockIdx.x * blockDim.x + threadIdx.x;
    if (idx >= Sm) return;
    int f = idx % Fm, p = idx / Fm;
    int b = p / HWm, rem = p % HWm, y = rem / Wm, x = rem % Wm;
    float s0 = 0.f, s2 = 0.f, s4 = 0.f;
#pragma unroll
    for (int ky = 0; ky < 3; ++ky) {
        int yy = y + ky - 1;
        if (yy < 0 || yy >= Hm) continue;
#pragma unroll
        for (int kx = 0; kx < 3; ++kx) {
            int xx = x + kx - 1;
            if (xx < 0 || xx >= Wm) continue;
            size_t pp = ((size_t)(b * Hm + yy) * Wm + xx) * Fm + f;
            float zv = z[pp], hv = hbuf[pp];
            const float* w = dw + ((size_t)(ky * 3 + kx) * Fm + f) * 6;
            s0 += zv * w[0] + hv * w[1];
            s2 += zv * w[2] + hv * w[3];
            s4 += zv * w[4] + hv * w[5];
        }
    }
    float i_s = hsig(s0 + sa_bias[f]);
    float g_s = tanhf(s2 + sa_bias[Fm + f]);
    float m   = (1.0f - i_s) * m_tm1[idx] + i_s * g_s;
    float o_s = hsig(s4 + sa_bias[2 * Fm + f]);
    out[2 * Sm + idx] = m;          // m
    out[idx]          = o_s * m;    // h_out
}

// ---------------- launcher ----------------
static inline size_t al256(size_t x) { return (x + 255) & ~(size_t)255; }

extern "C" void kernel_launch(void* const* d_in, const int* in_sizes, int n_in,
                              void* d_out, int out_size, void* d_ws, size_t ws_size,
                              hipStream_t stream) {
    (void)in_sizes; (void)n_in; (void)out_size; (void)ws_size;
    const float* inputs  = (const float*)d_in[0];
    const float* h_tm1   = (const float*)d_in[1];
    const float* c_tm1   = (const float*)d_in[2];
    const float* m_tm1   = (const float*)d_in[3];
    const float* kernel  = (const float*)d_in[4];
    const float* rkernel = (const float*)d_in[5];
    const float* bias    = (const float*)d_in[6];
    const float* sa_k    = (const float*)d_in[7];
    const float* kz      = (const float*)d_in[8];
    const float* dw      = (const float*)d_in[9];
    const float* sa_b    = (const float*)d_in[10];
    float* out = (float*)d_out;

    char* ws = (char*)d_ws;
    size_t off = 0;
    __bf16* xh  = (__bf16*)(ws + off); off = al256(off + (size_t)Mm * 192 * 2);
    __bf16* wqf = (__bf16*)(ws + off); off = al256(off + (size_t)KKm * NGm * 2);
    float*  g   = (float*) (ws + off); off = al256(off + (size_t)Mm * NGm * 4);
    float*  hb  = (float*) (ws + off); off = al256(off + (size_t)Sm * 4);
    float* qkv  = (float*) (ws + off); off = al256(off + (size_t)5 * Mm * 4);
    float*  z2  = (float*) (ws + off); off = al256(off + (size_t)Mm * 2 * 4);
    float*  z   = (float*) (ws + off); off = al256(off + (size_t)Sm * 4);

    pack_xh<<<(Mm * 192 + 255) / 256, 256, 0, stream>>>(inputs, h_tm1, xh);
    pack_w <<<(KKm * NGm + 255) / 256, 256, 0, stream>>>(kernel, rkernel, wqf);

    dim3 gg(Mm / 64, NGm / 192);
    gates_wmma<<<gg, 256, 0, stream>>>(xh, wqf, g);

    lstm_pointwise<<<(Sm + 255) / 256, 256, 0, stream>>>(g, bias, c_tm1, out, hb);

    qkv_proj<<<(5 * Mm + 127) / 128, 128, 0, stream>>>(hb, m_tm1, sa_k, qkv);

    dim3 ga(Bm, HWm / 64);
    sam_attention<<<ga, 64, 0, stream>>>(qkv, z2);

    conv_z<<<(Sm + 255) / 256, 256, 0, stream>>>(z2, kz, z);

    sam_update<<<(Sm + 255) / 256, 256, 0, stream>>>(z, hb, dw, sa_b, m_tm1, out);
}
